// GRANMixtureBernoulli_10015863734699
// MI455X (gfx1250) — matrix operand order repaired and checked by reading, not compiled, not run on hardware
//
#include <hip/hip_runtime.h>
#include <hip/hip_bf16.h>

typedef __attribute__((ext_vector_type(16))) _Float16 v16h;
typedef __attribute__((ext_vector_type(8)))  float    v8f;

namespace {
constexpr int Nn   = 50000;
constexpr int Me   = 250000;
constexpr int Eg   = 200000;
constexpr int MAXN = 512;
constexpr int H    = 128;
constexpr int EF   = 128;
constexpr int Kk   = 20;
constexpr int Ss   = 512;
constexpr int BC   = 16;
}

// ---------------- device helpers ----------------

__device__ __forceinline__ float4 ld4(const float* p) { return *reinterpret_cast<const float4*>(p); }
__device__ __forceinline__ float4 relu4(float4 v) {
  v.x = fmaxf(v.x, 0.f); v.y = fmaxf(v.y, 0.f); v.z = fmaxf(v.z, 0.f); v.w = fmaxf(v.w, 0.f);
  return v;
}
__device__ __forceinline__ float4 sub4(float4 a, float4 b) {
  float4 r; r.x = a.x - b.x; r.y = a.y - b.y; r.z = a.z - b.z; r.w = a.w - b.w; return r;
}

template <int OFF>
__device__ __forceinline__ void put8(v16h& a, float4 x, float4 y) {
  a[OFF + 0] = (_Float16)x.x; a[OFF + 1] = (_Float16)x.y;
  a[OFF + 2] = (_Float16)x.z; a[OFF + 3] = (_Float16)x.w;
  a[OFF + 4] = (_Float16)y.x; a[OFF + 5] = (_Float16)y.y;
  a[OFF + 6] = (_Float16)y.z; a[OFF + 7] = (_Float16)y.w;
}

union H16u { v16h v; float4 q[2]; };

__device__ __forceinline__ v8f vzero() {
  v8f z;
#pragma unroll
  for (int i = 0; i < 8; ++i) z[i] = 0.f;
  return z;
}

__device__ __forceinline__ v8f wmma_f16(v16h a, v16h b, v8f c) {
  return __builtin_amdgcn_wmma_f32_16x16x32_f16(false, a, false, b, (short)0, c, false, false);
}

// Native FP32 global atomic add (global_atomic_add_f32, no CAS loop)
__device__ __forceinline__ void gatomic_add(float* p, float v) {
  unsafeAtomicAdd(p, v);
}

// A fragment (16x32 f16 tile) from an fp32 row: runs at k0..k0+7 and k0+16..k0+23
__device__ __forceinline__ v16h a_frag_f32(const float* rowp, int k0, int relu) {
  float4 x0 = ld4(rowp + k0), x1 = ld4(rowp + k0 + 4);
  float4 y0 = ld4(rowp + k0 + 16), y1 = ld4(rowp + k0 + 20);
  if (relu) { x0 = relu4(x0); x1 = relu4(x1); y0 = relu4(y0); y1 = relu4(y1); }
  v16h a; put8<0>(a, x0, x1); put8<8>(a, y0, y1); return a;
}

// A fragment from an f16 row in LDS
__device__ __forceinline__ v16h a_frag_h(const _Float16* rowp, int k0) {
  H16u u;
  u.q[0] = *reinterpret_cast<const float4*>(rowp + k0);
  u.q[1] = *reinterpret_cast<const float4*>(rowp + k0 + 16);
  return u.v;
}

// B fragment: W is [Nout][ldk] row-major f16; lane (n = lane&15) needs runs at kbase+sel*8..+7 and +16..+23
__device__ __forceinline__ v16h b_frag(const _Float16* W, int ldk, int rowbase, int kbase, int lane) {
  int n = lane & 15, sel = lane >> 4;
  const _Float16* p = W + (size_t)(rowbase + n) * ldk + kbase + sel * 8;
  H16u u;
  u.q[0] = *reinterpret_cast<const float4*>(p);
  u.q[1] = *reinterpret_cast<const float4*>(p + 16);
  return u.v;
}

// A fragment for ei = concat(relu?(state[src]) - relu?(state[dst]), edge_feat) ; K = 256
__device__ __forceinline__ v16h ei_frag(const float* s, const float* d, const float* ef,
                                        int k0, int relu) {
  v16h a;
  if (k0 < H) {
    float4 x0 = ld4(s + k0), x1 = ld4(s + k0 + 4);
    float4 y0 = ld4(d + k0), y1 = ld4(d + k0 + 4);
    if (relu) { x0 = relu4(x0); x1 = relu4(x1); y0 = relu4(y0); y1 = relu4(y1); }
    put8<0>(a, sub4(x0, y0), sub4(x1, y1));
  } else {
    put8<0>(a, ld4(ef + k0 - H), ld4(ef + k0 - H + 4));
  }
  int k1 = k0 + 16;
  if (k1 < H) {
    float4 x0 = ld4(s + k1), x1 = ld4(s + k1 + 4);
    float4 y0 = ld4(d + k1), y1 = ld4(d + k1 + 4);
    if (relu) { x0 = relu4(x0); x1 = relu4(x1); y0 = relu4(y0); y1 = relu4(y1); }
    put8<8>(a, sub4(x0, y0), sub4(x1, y1));
  } else {
    put8<8>(a, ld4(ef + k1 - H), ld4(ef + k1 - H + 4));
  }
  return a;
}

__device__ __forceinline__ v16h diff_frag(const float* r0, const float* r1, int k0) {
  v16h a;
  put8<0>(a, sub4(ld4(r0 + k0), ld4(r1 + k0)), sub4(ld4(r0 + k0 + 4), ld4(r1 + k0 + 4)));
  put8<8>(a, sub4(ld4(r0 + k0 + 16), ld4(r1 + k0 + 16)), sub4(ld4(r0 + k0 + 20), ld4(r1 + k0 + 20)));
  return a;
}

__device__ __forceinline__ float sigmoidf_(float x) { return 1.f / (1.f + __expf(-x)); }
__device__ __forceinline__ float softplusf_(float t) {
  return fmaxf(t, 0.f) + log1pf(__expf(-fabsf(t)));
}

// hidden layer: out[16x128] = relu(A(16x128) @ W^T + b), A given as 4 k-chunk fragments
__device__ __forceinline__ void hidden_layer(const v16h* afr, const _Float16* W, const float* b,
                                             _Float16* tile_out, int m, int sel, int lane) {
  for (int nt = 0; nt < 8; ++nt) {
    v8f acc = vzero();
#pragma unroll
    for (int kc = 0; kc < 4; ++kc)
      acc = wmma_f16(afr[kc], b_frag(W, 128, nt * 16, kc * 32, lane), acc);
    float bn = b[nt * 16 + m];
#pragma unroll
    for (int r = 0; r < 8; ++r) {
      float v = acc[r] + bn;
      v = v > 0.f ? v : 0.f;
      tile_out[(r + sel * 8) * 128 + nt * 16 + m] = (_Float16)v;
    }
  }
}

// ---------------- utility kernels ----------------

__global__ void gran_cvt_f16(const float* __restrict__ s, _Float16* __restrict__ d, int n) {
  for (int i = blockIdx.x * blockDim.x + threadIdx.x; i < n; i += gridDim.x * blockDim.x)
    d[i] = (_Float16)s[i];
}
__global__ void gran_fill_f32(float* p, float v, int n) {
  for (int i = blockIdx.x * blockDim.x + threadIdx.x; i < n; i += gridDim.x * blockDim.x) p[i] = v;
}
__global__ void gran_fill_f16(_Float16* p, int n) {
  for (int i = blockIdx.x * blockDim.x + threadIdx.x; i < n; i += gridDim.x * blockDim.x)
    p[i] = (_Float16)0.f;
}

// ---------------- decoder: state = node_feat @ dec_W^T + dec_b ----------------

__global__ void gran_decoder(const float* __restrict__ nf, const _Float16* __restrict__ decW,
                             const float* __restrict__ decb, float* __restrict__ state) {
  int wave = threadIdx.x >> 5, lane = threadIdx.x & 31;
  int tile = blockIdx.x * 4 + wave;
  int rb = tile * 16;
  if (rb >= Nn) return;
  __builtin_prefetch(decW, 0, 3);
  int m = lane & 15, sel = lane >> 4;
  int row = rb + m; if (row >= Nn) row = Nn - 1;
  const float* arow = nf + (size_t)row * MAXN;
  v8f acc[8];
#pragma unroll
  for (int i = 0; i < 8; ++i) acc[i] = vzero();
  for (int kc = 0; kc < 16; ++kc) {   // K = 512
    int kb = kc * 32;
    v16h a = a_frag_f32(arow, kb + sel * 8, 0);
#pragma unroll
    for (int nt = 0; nt < 8; ++nt)
      acc[nt] = wmma_f16(a, b_frag(decW, MAXN, nt * 16, kb, lane), acc[nt]);
  }
#pragma unroll
  for (int nt = 0; nt < 8; ++nt) {
    int col = nt * 16 + m;
    float bn = decb[col];
#pragma unroll
    for (int r = 0; r < 8; ++r) {
      int rr = rb + r + sel * 8;
      if (rr < Nn) state[(size_t)rr * H + col] = acc[nt][r] + bn;
    }
  }
}

// ---------------- fused edge MLPs + attention + scatter into agg ----------------

__global__ void gran_edge(const float* __restrict__ state, const int* __restrict__ edge,
                          const float* __restrict__ efeat,
                          const _Float16* __restrict__ W1m, const float* __restrict__ b1m,
                          const _Float16* __restrict__ W2m, const float* __restrict__ b2m,
                          const _Float16* __restrict__ W1a, const float* __restrict__ b1a,
                          const _Float16* __restrict__ W2a, const float* __restrict__ b2a,
                          float* __restrict__ agg, int relu_state) {
  __shared__ _Float16 lds[4][2][16 * 128];
  int wave = threadIdx.x >> 5, lane = threadIdx.x & 31;
  long tile = blockIdx.x * 4 + wave;
  long eb = tile * 16;
  if (eb >= Me) return;
  __builtin_prefetch(W1m, 0, 3);
  __builtin_prefetch(W1a, 0, 3);
  __builtin_prefetch(W2m, 0, 3);
  __builtin_prefetch(W2a, 0, 3);
  int m = lane & 15, sel = lane >> 4;
  long e = eb + m; if (e >= Me) e = Me - 1;
  int sn = edge[2 * e], dn = edge[2 * e + 1];
  const float* srow = state + (size_t)sn * H;
  const float* drow = state + (size_t)dn * H;
  const float* erow = efeat + (size_t)e * EF;

  // stage 1: both hidden layers (K = 256 -> 128) in one pass over the gathered A
  // fragments (one ei gather feeds both the msg and att weight streams).
  {
    v8f accM[8], accA[8];
#pragma unroll
    for (int i = 0; i < 8; ++i) { accM[i] = vzero(); accA[i] = vzero(); }
    for (int kc = 0; kc < 8; ++kc) {   // K = 256
      int kb = kc * 32;
      v16h a = ei_frag(srow, drow, erow, kb + sel * 8, relu_state);
#pragma unroll
      for (int nt = 0; nt < 8; ++nt)
        accM[nt] = wmma_f16(a, b_frag(W1m, 256, nt * 16, kb, lane), accM[nt]);
#pragma unroll
      for (int nt = 0; nt < 8; ++nt)
        accA[nt] = wmma_f16(a, b_frag(W1a, 256, nt * 16, kb, lane), accA[nt]);
    }
#pragma unroll
    for (int nt = 0; nt < 8; ++nt) {
      float bm = b1m[nt * 16 + m];
      float ba = b1a[nt * 16 + m];
#pragma unroll
      for (int r = 0; r < 8; ++r) {
        float vm = accM[nt][r] + bm; vm = vm > 0.f ? vm : 0.f;
        float va = accA[nt][r] + ba; va = va > 0.f ? va : 0.f;
        lds[wave][0][(r + sel * 8) * 128 + nt * 16 + m] = (_Float16)vm;
        lds[wave][1][(r + sel * 8) * 128 + nt * 16 + m] = (_Float16)va;
      }
    }
  }

  // dst index per output row (C-fragment row mapping)
  int dstr[8];
#pragma unroll
  for (int r = 0; r < 8; ++r) {
    long e2 = eb + r + sel * 8; if (e2 >= Me) e2 = Me - 1;
    dstr[r] = edge[2 * e2 + 1];
  }

  // stage 2: msg = h1m @ W2m^T + b ; att = sigmoid(h1a @ W2a^T + b) ; scatter msg*att
  v16h aM[4], aA[4];
#pragma unroll
  for (int kc = 0; kc < 4; ++kc) {
    aM[kc] = a_frag_h(lds[wave][0] + m * 128, kc * 32 + sel * 8);
    aA[kc] = a_frag_h(lds[wave][1] + m * 128, kc * 32 + sel * 8);
  }
  for (int nt = 0; nt < 8; ++nt) {
    v8f am = vzero(), aa = vzero();
#pragma unroll
    for (int kc = 0; kc < 4; ++kc) {
      int kb = kc * 32;
      am = wmma_f16(aM[kc], b_frag(W2m, 128, nt * 16, kb, lane), am);
      aa = wmma_f16(aA[kc], b_frag(W2a, 128, nt * 16, kb, lane), aa);
    }
    int col = nt * 16 + m;
    float bm = b2m[col], ba = b2a[col];
#pragma unroll
    for (int r = 0; r < 8; ++r) {
      long e2 = eb + r + sel * 8;
      if (e2 < Me) {
        float msg = am[r] + bm;
        float att = sigmoidf_(aa[r] + ba);
        gatomic_add(&agg[(size_t)dstr[r] * H + col], msg * att);
      }
    }
  }
}

// ---------------- fused GRU update (in-place on state) ----------------

__global__ void gran_gru(const float* __restrict__ agg, float* __restrict__ state,
                         const _Float16* __restrict__ Wih, const float* __restrict__ bih,
                         const _Float16* __restrict__ Whh, const float* __restrict__ bhh,
                         int relu_state) {
  int wave = threadIdx.x >> 5, lane = threadIdx.x & 31;
  int tile = blockIdx.x * 4 + wave;
  int rb = tile * 16;
  if (rb >= Nn) return;
  __builtin_prefetch(Wih, 0, 3);
  __builtin_prefetch(Whh, 0, 3);
  int m = lane & 15, sel = lane >> 4;
  int row = rb + m; if (row >= Nn) row = Nn - 1;
  const float* arow = agg + (size_t)row * H;
  const float* hrow = state + (size_t)row * H;
  v16h aA[4], aH[4];
#pragma unroll
  for (int kc = 0; kc < 4; ++kc) {
    aA[kc] = a_frag_f32(arow, kc * 32 + sel * 8, 0);
    aH[kc] = a_frag_f32(hrow, kc * 32 + sel * 8, relu_state);
  }
  for (int nt = 0; nt < 8; ++nt) {
    v8f ir = vzero(), iz = vzero(), in2 = vzero();
    v8f hr = vzero(), hz = vzero(), hn = vzero();
#pragma unroll
    for (int kc = 0; kc < 4; ++kc) {
      int kb = kc * 32;
      ir  = wmma_f16(aA[kc], b_frag(Wih, 128,       nt * 16, kb, lane), ir);
      iz  = wmma_f16(aA[kc], b_frag(Wih, 128, 128 + nt * 16, kb, lane), iz);
      in2 = wmma_f16(aA[kc], b_frag(Wih, 128, 256 + nt * 16, kb, lane), in2);
      hr  = wmma_f16(aH[kc], b_frag(Whh, 128,       nt * 16, kb, lane), hr);
      hz  = wmma_f16(aH[kc], b_frag(Whh, 128, 128 + nt * 16, kb, lane), hz);
      hn  = wmma_f16(aH[kc], b_frag(Whh, 128, 256 + nt * 16, kb, lane), hn);
    }
    int col = nt * 16 + m;
    float bir = bih[col], biz = bih[128 + col], bin = bih[256 + col];
    float bhr = bhh[col], bhz = bhh[128 + col], bhn = bhh[256 + col];
#pragma unroll
    for (int r = 0; r < 8; ++r) {
      int rr = rb + r + sel * 8;
      if (rr < Nn) {
        float st = state[(size_t)rr * H + col];
        if (relu_state) st = st > 0.f ? st : 0.f;
        float rg = sigmoidf_(ir[r] + bir + hr[r] + bhr);
        float zg = sigmoidf_(iz[r] + biz + hz[r] + bhz);
        float ng = tanhf(in2[r] + bin + rg * (hn[r] + bhn));
        state[(size_t)rr * H + col] = (1.f - zg) * ng + zg * st;
      }
    }
  }
}

// ---------------- segment counts ----------------

__global__ void gran_segcount(const int* __restrict__ sg, float* __restrict__ cnt) {
  int i = blockIdx.x * blockDim.x + threadIdx.x;
  if (i < Eg) gatomic_add(&cnt[sg[i]], 1.0f);
}

// ---------------- fused output MLPs + loss accumulation ----------------

__global__ void gran_outloss(const float* __restrict__ state, const int* __restrict__ nidx,
                             const float* __restrict__ label, const int* __restrict__ sgidx,
                             const _Float16* __restrict__ tW1, const float* __restrict__ tb1,
                             const _Float16* __restrict__ tW2, const float* __restrict__ tb2,
                             const _Float16* __restrict__ tW3, const float* __restrict__ tb3,
                             const _Float16* __restrict__ aW1, const float* __restrict__ ab1,
                             const _Float16* __restrict__ aW2, const float* __restrict__ ab2,
                             const _Float16* __restrict__ aW3, const float* __restrict__ ab3,
                             float* __restrict__ rAdj, float* __restrict__ rLa) {
  __shared__ _Float16 lds[4][2][16 * 128];
  int wave = threadIdx.x >> 5, lane = threadIdx.x & 31;
  long tile = blockIdx.x * 4 + wave;
  long eb = tile * 16;
  if (eb >= Eg) return;
  __builtin_prefetch(tW1, 0, 3);
  __builtin_prefetch(aW1, 0, 3);
  int m = lane & 15, sel = lane >> 4;
  long e = eb + m; if (e >= Eg) e = Eg - 1;
  int i0 = nidx[2 * e], i1 = nidx[2 * e + 1];
  const float* r0 = state + (size_t)i0 * H;
  const float* r1 = state + (size_t)i1 * H;

  v16h aD[4];
#pragma unroll
  for (int kc = 0; kc < 4; ++kc) aD[kc] = diff_frag(r0, r1, kc * 32 + sel * 8);

  float yv[8]; int sg[8]; int ok[8];
#pragma unroll
  for (int r = 0; r < 8; ++r) {
    long e2 = eb + r + sel * 8;
    ok[r] = (e2 < Eg);
    long ec = ok[r] ? e2 : (Eg - 1);
    yv[r] = label[ec];
    sg[r] = sgidx[ec];
  }

  for (int path = 0; path < 2; ++path) {
    const _Float16* W1 = path ? aW1 : tW1; const float* b1 = path ? ab1 : tb1;
    const _Float16* W2 = path ? aW2 : tW2; const float* b2 = path ? ab2 : tb2;
    const _Float16* W3 = path ? aW3 : tW3; const float* b3 = path ? ab3 : tb3;
    float* racc = path ? rLa : rAdj;

    hidden_layer(aD, W1, b1, lds[wave][0], m, sel, lane);
    v16h a1[4];
#pragma unroll
    for (int kc = 0; kc < 4; ++kc) a1[kc] = a_frag_h(lds[wave][0] + m * 128, kc * 32 + sel * 8);
    hidden_layer(a1, W2, b2, lds[wave][1], m, sel, lane);
    v16h a2[4];
#pragma unroll
    for (int kc = 0; kc < 4; ++kc) a2[kc] = a_frag_h(lds[wave][1] + m * 128, kc * 32 + sel * 8);

    for (int nt = 0; nt < 2; ++nt) {   // Nout padded to 32; real cols < 20
      v8f acc = vzero();
#pragma unroll
      for (int kc = 0; kc < 4; ++kc)
        acc = wmma_f16(a2[kc], b_frag(W3, 128, nt * 16, kc * 32, lane), acc);
      int col = nt * 16 + m;
      if (col < Kk) {
        float b3v = b3[col];
#pragma unroll
        for (int r = 0; r < 8; ++r) {
          if (ok[r]) {
            float v = acc[r] + b3v;
            float contrib = (path == 0) ? (softplusf_(-v) + (1.f - yv[r]) * v) : v;
            gatomic_add(&racc[sg[r] * Kk + col], contrib);
          }
        }
      }
    }
  }
}

// ---------------- final reduction to scalar loss ----------------

__global__ void gran_finalize(const float* __restrict__ rAdj, const float* __restrict__ rLa,
                              const float* __restrict__ cnt, const int* __restrict__ bcidx,
                              float* __restrict__ out) {
  __shared__ float bcp[BC], bcc[BC];
  int s = threadIdx.x;
  if (s < BC) { bcp[s] = 0.f; bcc[s] = 0.f; }
  __syncthreads();
  float c = cnt[s];
  float la[Kk];
  float mx = -3.0e38f;
  for (int k = 0; k < Kk; ++k) { la[k] = rLa[s * Kk + k] / c; mx = fmaxf(mx, la[k]); }
  float sum = 0.f;
  for (int k = 0; k < Kk; ++k) sum += expf(la[k] - mx);
  float lse = mx + logf(sum);
  float mx2 = -3.0e38f;
  float vv[Kk];
  for (int k = 0; k < Kk; ++k) { vv[k] = -rAdj[s * Kk + k] + (la[k] - lse); mx2 = fmaxf(mx2, vv[k]); }
  sum = 0.f;
  for (int k = 0; k < Kk; ++k) sum += expf(vv[k] - mx2);
  float lp = mx2 + logf(sum);
  int b = bcidx[s];
  atomicAdd(&bcp[b], lp);
  atomicAdd(&bcc[b], c);
  __syncthreads();
  if (s == 0) {
    float acc = 0.f;
    for (int i = 0; i < BC; ++i) acc += bcp[i] / bcc[i];   // C == 1 -> logsumexp is identity
    out[0] = -acc / (float)BC;
  }
}

// ---------------- host launch ----------------

extern "C" void kernel_launch(void* const* d_in, const int* in_sizes, int n_in,
                              void* d_out, int out_size, void* d_ws, size_t ws_size,
                              hipStream_t stream) {
  (void)in_sizes; (void)n_in; (void)out_size; (void)ws_size;
  const float* node_feat = (const float*)d_in[0];
  const int*   edge      = (const int*)d_in[1];
  const float* edge_feat = (const float*)d_in[2];
  const int*   nidx      = (const int*)d_in[3];
  const float* label     = (const float*)d_in[4];
  const int*   sgidx     = (const int*)d_in[5];
  const int*   bcidx     = (const int*)d_in[6];
  const float* dec_W  = (const float*)d_in[7];  const float* dec_b  = (const float*)d_in[8];
  const float* msg_W1 = (const float*)d_in[9];  const float* msg_b1 = (const float*)d_in[10];
  const float* msg_W2 = (const float*)d_in[11]; const float* msg_b2 = (const float*)d_in[12];
  const float* att_W1 = (const float*)d_in[13]; const float* att_b1 = (const float*)d_in[14];
  const float* att_W2 = (const float*)d_in[15]; const float* att_b2 = (const float*)d_in[16];
  const float* gru_Wih = (const float*)d_in[17]; const float* gru_bih = (const float*)d_in[18];
  const float* gru_Whh = (const float*)d_in[19]; const float* gru_bhh = (const float*)d_in[20];
  const float* th_W1 = (const float*)d_in[21]; const float* th_b1 = (const float*)d_in[22];
  const float* th_W2 = (const float*)d_in[23]; const float* th_b2 = (const float*)d_in[24];
  const float* th_W3 = (const float*)d_in[25]; const float* th_b3 = (const float*)d_in[26];
  const float* al_W1 = (const float*)d_in[27]; const float* al_b1 = (const float*)d_in[28];
  const float* al_W2 = (const float*)d_in[29]; const float* al_b2 = (const float*)d_in[30];
  const float* al_W3 = (const float*)d_in[31]; const float* al_b3 = (const float*)d_in[32];

  // ---- workspace layout ----
  float* state = (float*)d_ws;                 // Nn*H
  float* agg   = state + (size_t)Nn * H;       // Nn*H
  float* rAdj  = agg + (size_t)Nn * H;         // Ss*Kk
  float* rLa   = rAdj + Ss * Kk;               // Ss*Kk
  float* cnt   = rLa + Ss * Kk;                // Ss
  size_t fbytes = (size_t)((char*)(cnt + Ss) - (char*)d_ws);
  fbytes = (fbytes + 255) & ~(size_t)255;
  _Float16* wh = (_Float16*)((char*)d_ws + fbytes);

  _Float16* hDec   = wh;                 // 65536
  _Float16* hMsgW1 = hDec + 65536;       // 3*32768
  _Float16* hAttW1 = hMsgW1 + 98304;     // 3*32768
  _Float16* hMsgW2 = hAttW1 + 98304;     // 3*16384
  _Float16* hAttW2 = hMsgW2 + 49152;     // 3*16384
  _Float16* hWih   = hAttW2 + 49152;     // 3*49152
  _Float16* hWhh   = hWih + 147456;      // 3*49152
  _Float16* hThW1  = hWhh + 147456;      // 16384
  _Float16* hThW2  = hThW1 + 16384;
  _Float16* hAlW1  = hThW2 + 16384;
  _Float16* hAlW2  = hAlW1 + 16384;
  _Float16* hThW3  = hAlW2 + 16384;      // 32x128 zero-padded
  _Float16* hAlW3  = hThW3 + 4096;

  auto cvt = [&](const float* src, _Float16* dst, int n) {
    int blocks = (n + 255) / 256; if (blocks > 4096) blocks = 4096;
    gran_cvt_f16<<<blocks, 256, 0, stream>>>(src, dst, n);
  };

  gran_fill_f16<<<32, 256, 0, stream>>>(hThW3, 8192);  // zero both padded W3s (contiguous)
  cvt(dec_W, hDec, 65536);
  cvt(msg_W1, hMsgW1, 98304);  cvt(att_W1, hAttW1, 98304);
  cvt(msg_W2, hMsgW2, 49152);  cvt(att_W2, hAttW2, 49152);
  cvt(gru_Wih, hWih, 147456);  cvt(gru_Whh, hWhh, 147456);
  cvt(th_W1, hThW1, 16384);    cvt(th_W2, hThW2, 16384);
  cvt(al_W1, hAlW1, 16384);    cvt(al_W2, hAlW2, 16384);
  cvt(th_W3, hThW3, Kk * H);   cvt(al_W3, hAlW3, Kk * H);

  const int nodeBlocks = (((Nn + 15) / 16) + 3) / 4;   // 782
  const int edgeBlocks = (((Me + 15) / 16) + 3) / 4;   // 3907
  const int egBlocks   = (((Eg + 15) / 16) + 3) / 4;   // 3125

  gran_decoder<<<nodeBlocks, 128, 0, stream>>>(node_feat, hDec, dec_b, state);

  for (int l = 0; l < 3; ++l) {
    gran_fill_f32<<<2048, 256, 0, stream>>>(agg, 0.f, Nn * H);
    gran_edge<<<edgeBlocks, 128, 0, stream>>>(
        state, edge, edge_feat,
        hMsgW1 + (size_t)l * 32768, msg_b1 + l * 128, hMsgW2 + (size_t)l * 16384, msg_b2 + l * 128,
        hAttW1 + (size_t)l * 32768, att_b1 + l * 128, hAttW2 + (size_t)l * 16384, att_b2 + l * 128,
        agg, l > 0 ? 1 : 0);
    gran_gru<<<nodeBlocks, 128, 0, stream>>>(
        agg, state, hWih + (size_t)l * 49152, gru_bih + l * 384,
        hWhh + (size_t)l * 49152, gru_bhh + l * 384, l > 0 ? 1 : 0);
  }

  gran_fill_f32<<<96, 256, 0, stream>>>(rAdj, 0.f, Ss * Kk * 2 + Ss);  // rAdj, rLa, cnt contiguous
  gran_segcount<<<(Eg + 255) / 256, 256, 0, stream>>>(sgidx, cnt);
  gran_outloss<<<egBlocks, 128, 0, stream>>>(
      state, nidx, label, sgidx,
      hThW1, th_b1, hThW2, th_b2, hThW3, th_b3,
      hAlW1, al_b1, hAlW2, al_b2, hAlW3, al_b3,
      rAdj, rLa);
  gran_finalize<<<1, Ss, 0, stream>>>(rAdj, rLa, cnt, bcidx, (float*)d_out);
}
